// ImplicitEmbedding_42236708389222
// MI455X (gfx1250) — compile-verified, matrix operand order checked by reading, not run on hardware
//
#include <hip/hip_runtime.h>
#include <hip/hip_bf16.h>

typedef __attribute__((ext_vector_type(16))) _Float16 v16h;
typedef __attribute__((ext_vector_type(8)))  float    v8f;

#define OMEGA0      30.0f
#define TOK_PER_BLK 128
#define THREADS     256
#define MTILES      8     // 128 tokens / 16 rows per tile
#define FRAG_DW     (2 * 32 * 8)          // dwords per mtile (2 kb regions)
#define XBUF_DW     (MTILES * FRAG_DW)    // 4096 dwords = 16 KB

// 16-bit matrix fragment half-index -> K mapping (per CDNA5 ISA A-layout):
// lanes 0-15 (hi=0): halves 0..7 -> K 0..7, halves 8..15 -> K 16..23
// lanes 16-31 (hi=1): same +8
__device__ __forceinline__ int kmap(int h, int hi) {
  return (h < 8 ? h : h + 8) + (hi ? 8 : 0);
}

// Build a B fragment (K x 16 slice of W, row-major W[K][ldn]) as f16 from f32 global.
__device__ __forceinline__ v16h load_bfrag(const float* __restrict__ W, int ldn,
                                           int Kdim, int kbase, int n0, int lane) {
  int n  = n0 + (lane & 15);
  int hi = lane >> 4;
  v16h b;
#pragma unroll
  for (int h = 0; h < 16; ++h) {
    int k = kbase + kmap(h, hi);
    float v = 0.0f;
    if (k < Kdim) v = W[k * ldn + n];   // EXEC-masked: never reads OOB
    b[h] = (_Float16)v;
  }
  return b;
}

// A fragments live in LDS in fragment order: region(mtile,kb) is 32 lanes x 32B.
__device__ __forceinline__ v16h load_afrag(const unsigned int* sX, int mtile, int kb, int lane) {
  return *(const v16h*)(sX + (((mtile * 2 + kb) * 32 + lane) << 3));
}

// Apply bias + sin, convert to f16, scatter accumulator into A-fragment layout
// for the next layer (output column n becomes next-layer K index).
__device__ __forceinline__ void store_act(unsigned int* dst, int mtile, int ntile,
                                          v8f c, const float* __restrict__ bias, int lane) {
  _Float16* sH = (_Float16*)dst;
  int l   = lane & 15;
  int chi = lane >> 4;
  int kk  = ntile * 16 + l;            // next-layer K, 0..63
  float bv = bias[kk];
  int kb   = kk >> 5;
  int kr   = kk & 31;
  int hi_t = (kr >> 3) & 1;            // inverse of kmap
  int h    = (kr & 7) + ((kr >> 4) << 3);
  int base = ((mtile * 2 + kb) * 32) << 4;   // half-index of region start
#pragma unroll
  for (int r = 0; r < 8; ++r) {
    int m      = r + chi * 8;          // row within M-tile (C layout)
    int lane_t = m + hi_t * 16;
    float x = __sinf(OMEGA0 * (c[r] + bv));
    sH[base + lane_t * 16 + h] = (_Float16)x;
  }
}

// One hidden layer, src -> dst (ping-pong LDS buffers).
// Wave w owns N-tile (w>>1) and M-tiles (w&1)*4 .. +3, so each W fragment is
// loaded once per wave and reused across 4 M-tiles.
__device__ __forceinline__ void hidden_layer(const unsigned int* src, unsigned int* dst,
                                             const float* __restrict__ W,
                                             const float* __restrict__ bias, int Kdim,
                                             int wave, int lane) {
  int nt  = wave >> 1;
  int mt0 = (wave & 1) * 4;
  v16h b0 = load_bfrag(W, 64, Kdim, 0, nt * 16, lane);
  v16h b1 = {};
  if (Kdim > 32) b1 = load_bfrag(W, 64, Kdim, 32, nt * 16, lane);
#pragma unroll
  for (int i = 0; i < 4; ++i) {
    int mt = mt0 + i;
    v16h a0 = load_afrag(src, mt, 0, lane);
    v8f c = {};
    c = __builtin_amdgcn_wmma_f32_16x16x32_f16(false, a0, false, b0, (short)0, c, false, false);
    if (Kdim > 32) {
      v16h a1 = load_afrag(src, mt, 1, lane);
      c = __builtin_amdgcn_wmma_f32_16x16x32_f16(false, a1, false, b1, (short)0, c, false, false);
    }
    store_act(dst, mt, nt, c, bias, lane);
  }
  __syncthreads();
}

// Output projection 64 -> 1024. Wave w owns N-tiles w*8 .. w*8+7 and sweeps all
// 8 M-tiles, so each Wout fragment pair is built once and reused 8x.
// Output (256 MB, write-once, never re-read, > 192 MB L2) uses non-temporal
// stores so the streaming traffic doesn't evict Wout/coords from L2.
__device__ __forceinline__ void out_layer(const unsigned int* sX,
                                          const float* __restrict__ Wout,
                                          const float* __restrict__ bout,
                                          float* __restrict__ out,
                                          int token0, int wave, int lane) {
  int l  = lane & 15;
  int hi = lane >> 4;
  for (int i = 0; i < 8; ++i) {
    int nt = wave * 8 + i;
    int n  = nt * 16 + l;
    float bo = bout[n];
    v16h b0 = load_bfrag(Wout, 1024, 64, 0,  nt * 16, lane);
    v16h b1 = load_bfrag(Wout, 1024, 64, 32, nt * 16, lane);
#pragma unroll
    for (int mt = 0; mt < 8; ++mt) {
      v16h a0 = load_afrag(sX, mt, 0, lane);
      v16h a1 = load_afrag(sX, mt, 1, lane);
      v8f c;
#pragma unroll
      for (int r = 0; r < 8; ++r) c[r] = bo;   // C init = bias
      c = __builtin_amdgcn_wmma_f32_16x16x32_f16(false, a0, false, b0, (short)0, c, false, false);
      c = __builtin_amdgcn_wmma_f32_16x16x32_f16(false, a1, false, b1, (short)0, c, false, false);
      size_t row0 = (size_t)token0 + mt * 16 + hi * 8;
#pragma unroll
      for (int r = 0; r < 8; ++r)
        __builtin_nontemporal_store(c[r], &out[(row0 + r) * 1024 + n]);
    }
  }
}

__global__ __launch_bounds__(THREADS)
void ImplicitEmbedding_42236708389222_kernel(
    const int*   __restrict__ ids,   const float* __restrict__ coords,
    const float* __restrict__ W0,    const float* __restrict__ b0,
    const float* __restrict__ W1,    const float* __restrict__ b1,
    const float* __restrict__ W2,    const float* __restrict__ b2,
    const float* __restrict__ Wout,  const float* __restrict__ bout,
    float* __restrict__ out) {
  __shared__ __align__(32) unsigned int sXa[XBUF_DW];  // 16 KB
  __shared__ __align__(32) unsigned int sXb[XBUF_DW];  // 16 KB

  int tid    = threadIdx.x;
  int lane   = tid & 31;
  int wave   = tid >> 5;
  int token0 = blockIdx.x * TOK_PER_BLK;

  // Phase 0: gather coords -> X0 A-fragments in sXa (f16, K padded 16 -> 32)
  if (tid < TOK_PER_BLK) {
    int id = ids[token0 + tid];
    const float* cp = coords + (size_t)id * 16;
    float c16[16];
#pragma unroll
    for (int k = 0; k < 16; ++k) c16[k] = cp[k];
    int mtile = tid >> 4, mrow = tid & 15;
#pragma unroll
    for (int hi = 0; hi < 2; ++hi) {
      unsigned int* p = sXa + (((mtile * 2 + 0) * 32 + (mrow + hi * 16)) << 3);
#pragma unroll
      for (int v = 0; v < 8; ++v) {
        int k0 = kmap(2 * v, hi), k1 = kmap(2 * v + 1, hi);
        float f0 = (k0 < 16) ? c16[k0 < 16 ? k0 : 0] : 0.0f;
        float f1 = (k1 < 16) ? c16[k1 < 16 ? k1 : 0] : 0.0f;
        union { _Float16 hh[2]; unsigned int u; } pk;
        pk.hh[0] = (_Float16)f0; pk.hh[1] = (_Float16)f1;
        p[v] = pk.u;
      }
    }
  }
  __syncthreads();

  hidden_layer(sXa, sXb, W0, b0, 16, wave, lane);   // sXa -> sXb
  hidden_layer(sXb, sXa, W1, b1, 64, wave, lane);   // sXb -> sXa
  hidden_layer(sXa, sXb, W2, b2, 64, wave, lane);   // sXa -> sXb
  out_layer(sXb, Wout, bout, out, token0, wave, lane);
}

extern "C" void kernel_launch(void* const* d_in, const int* in_sizes, int n_in,
                              void* d_out, int out_size, void* d_ws, size_t ws_size,
                              hipStream_t stream) {
  const int*   ids    = (const int*)  d_in[0];
  const float* coords = (const float*)d_in[1];
  const float* W0     = (const float*)d_in[2];
  const float* b0     = (const float*)d_in[3];
  const float* W1     = (const float*)d_in[4];
  const float* b1     = (const float*)d_in[5];
  const float* W2     = (const float*)d_in[6];
  const float* b2     = (const float*)d_in[7];
  const float* Wout   = (const float*)d_in[8];
  const float* bout   = (const float*)d_in[9];
  float* out = (float*)d_out;

  int nTok   = in_sizes[0];                       // 16 * 4096 = 65536
  int blocks = (nTok + TOK_PER_BLK - 1) / TOK_PER_BLK;
  ImplicitEmbedding_42236708389222_kernel<<<blocks, THREADS, 0, stream>>>(
      ids, coords, W0, b0, W1, b1, W2, b2, Wout, bout, out);
}
